// FrequencyAwareGQA_46291157517075
// MI455X (gfx1250) — matrix operand order repaired and checked by reading, not compiled, not run on hardware
//
#include <hip/hip_runtime.h>
#include <hip/hip_bf16.h>

// ---------------------------------------------------------------------------
// FrequencyAwareGQA for MI455X (gfx1250, wave32, WMMA 16x16x32 f16,
// async-to-LDS staging).
//   1. f32->f16 converts of x, Wq, Wkv, Wo
//   2. WMMA GEMM (async double-buffered): q_raw = x@Wq, kv_raw = x@Wkv
//   3. LayerNorm(q) + atomic per-(b,h,d) token sums
//   4. q finalize: mean subtract + RoPE -> f16 (B,H,T,D)
//   5. kv prep: RoPE(k), layout -> f16 (B,KVH,T,D)
//   6. Flash attention: 64x64 tiles, online softmax, async V staging
//   7. WMMA GEMM: out = att@Wo -> f32 d_out
// ---------------------------------------------------------------------------

constexpr int B_  = 2;
constexpr int T_  = 2048;
constexpr int C_  = 1024;
constexpr int H_  = 16;
constexpr int D_  = 64;
constexpr int KVH_= 3;
constexpr float SCALE_ = 0.125f;     // D^-0.5
constexpr float LN_EPS_ = 1e-5f;

typedef __attribute__((ext_vector_type(16))) _Float16 v16h;
typedef __attribute__((ext_vector_type(8)))  float    v8f;
typedef __attribute__((ext_vector_type(8)))  _Float16 h8;
typedef __attribute__((ext_vector_type(4)))  _Float16 h4;
typedef __attribute__((ext_vector_type(2)))  _Float16 h2;
typedef __attribute__((ext_vector_type(4)))  int      v4i;

// ---- CDNA5 async global->LDS copy (ASYNCcnt path) --------------------------
__device__ __forceinline__ void async_copy16(void* lds, const void* g) {
#if __has_builtin(__builtin_amdgcn_global_load_async_to_lds_b128)
  // builtin signature (from hipcc diagnostic): (AS1 v4i*, AS3 v4i*, imm, cpol)
  __builtin_amdgcn_global_load_async_to_lds_b128(
      (__attribute__((address_space(1))) v4i*)g,
      (__attribute__((address_space(3))) v4i*)lds, 0, 0);
#else
  asm volatile("global_load_async_to_lds_b128 %0, %1, off"
               :: "v"((unsigned)(uintptr_t)lds), "v"(g)
               : "memory");
#endif
}

__device__ __forceinline__ void wait_async0() {
#if __has_builtin(__builtin_amdgcn_s_wait_asynccnt)
  __builtin_amdgcn_s_wait_asynccnt(0);
#else
  asm volatile("s_wait_asynccnt 0x0" ::: "memory");
#endif
}

// ---- WMMA fragment loaders (layouts per CDNA5 ISA 7.12.2, wave32) ----------
// A (16x32 f16): lane L holds row M=L&15; VGPR r pairs: K=(r<4?2r:8+2r)+(L>=16?8:0)
__device__ __forceinline__ v16h wmma_frag_a(const _Float16* base, int ld) {
  const int lane = threadIdx.x & 31;
  const int m    = lane & 15;
  const int ko   = (lane & 16) ? 8 : 0;
  const _Float16* row = base + (size_t)m * ld;
  v16h a;
#pragma unroll
  for (int r = 0; r < 8; ++r) {
    const int k = ((r < 4) ? (2 * r) : (8 + 2 * r)) + ko;
    h2 p = *(const h2*)(row + k);
    a[2 * r]     = p[0];
    a[2 * r + 1] = p[1];
  }
  return a;
}

// B (32x16 f16): lane L holds column N=L&15; element s: K=k0+(s<8?s:8+s)+(L>=16?8:0).
__device__ __forceinline__ v16h wmma_frag_b(const _Float16* base, int ld, int k0, int n0) {
  const int lane = threadIdx.x & 31;
  const int n    = lane & 15;
  const int ko   = (lane & 16) ? 8 : 0;
  v16h b;
#pragma unroll
  for (int s = 0; s < 16; ++s) {
    const int k = ((s < 8) ? s : (8 + s)) + ko;
    b[s] = base[(size_t)(k0 + k) * ld + n0 + n];
  }
  return b;
}

__device__ __forceinline__ v8f wmma_mac(v16h a, v16h b, v8f c) {
  return __builtin_amdgcn_wmma_f32_16x16x32_f16(false, a, false, b, (short)0, c,
                                                false, false);
}

// ---- generic elementwise f32 -> f16 ---------------------------------------
__global__ void f32_to_f16_kernel(const float* __restrict__ in,
                                  _Float16* __restrict__ out, int n) {
  int i = blockIdx.x * 256 + threadIdx.x;
  if (i < n) out[i] = (_Float16)in[i];
}

// ---- tiled WMMA GEMM: C(f32, MxN) = A(f16, MxK) * B(f16, KxN) -------------
// 128 threads (4 waves), tile 128x64, K-chunks of 64, async double-buffered
// staging (one barrier per chunk). Each wave owns two 16-row strips and
// reuses every B fragment twice.
__global__ __launch_bounds__(128) void gemm_f16_f32(const _Float16* __restrict__ A,
                                                    const _Float16* __restrict__ Bm,
                                                    float* __restrict__ C,
                                                    int M, int N, int K) {
  __shared__ alignas(32) _Float16 As[2][128][64];
  __shared__ alignas(32) _Float16 Bs[2][64][64];
  const int m0   = blockIdx.x * 128;
  const int n0   = blockIdx.y * 64;
  const int tid  = threadIdx.x;
  const int w    = tid >> 5;
  const int lane = tid & 31;

  v8f acc[2][4];
#pragma unroll
  for (int s = 0; s < 2; ++s)
#pragma unroll
    for (int i = 0; i < 4; ++i)
#pragma unroll
      for (int j = 0; j < 8; ++j) acc[s][i][j] = 0.0f;

  auto stage = [&](int buf, int k0) {
#pragma unroll
    for (int i = 0; i < 8; ++i) {           // A: 1024 16B chunks / 128 thr
      const int o = tid + i * 128;
      const int row = o >> 3;
      const int c8  = (o & 7) * 8;
      async_copy16(&As[buf][row][c8], A + (size_t)(m0 + row) * K + k0 + c8);
    }
#pragma unroll
    for (int i = 0; i < 4; ++i) {           // B: 512 16B chunks / 128 thr
      const int o = tid + i * 128;
      const int row = o >> 3;
      const int c8  = (o & 7) * 8;
      async_copy16(&Bs[buf][row][c8], Bm + (size_t)(k0 + row) * N + n0 + c8);
    }
  };

  const int iters = K / 64;
  stage(0, 0);
  for (int it = 0; it < iters; ++it) {
    wait_async0();
    __syncthreads();                        // buf[it&1] published to all waves
    if (it + 1 < iters) stage((it + 1) & 1, (it + 1) * 64);
    const int buf = it & 1;
#pragma unroll
    for (int kc = 0; kc < 64; kc += 32) {
      v16h a0 = wmma_frag_a(&As[buf][w * 32][kc], 64);
      v16h a1 = wmma_frag_a(&As[buf][w * 32 + 16][kc], 64);
      v16h vb[4];
#pragma unroll
      for (int nt = 0; nt < 4; ++nt)
        vb[nt] = wmma_frag_b(&Bs[buf][0][0], 64, kc, nt * 16);
#pragma unroll
      for (int nt = 0; nt < 4; ++nt) {
        acc[0][nt] = wmma_mac(a0, vb[nt], acc[0][nt]);
        acc[1][nt] = wmma_mac(a1, vb[nt], acc[1][nt]);
      }
    }
  }

  const int hi = (lane >> 4) & 1;
  const int nl = lane & 15;
#pragma unroll
  for (int s = 0; s < 2; ++s)
#pragma unroll
    for (int nt = 0; nt < 4; ++nt)
#pragma unroll
      for (int r = 0; r < 8; ++r)
        C[(size_t)(m0 + w * 32 + s * 16 + r + hi * 8) * N + n0 + nt * 16 + nl] =
            acc[s][nt][r];
}

// ---- per-(b,t,h) LayerNorm over D=64 + token-sum accumulation --------------
__global__ __launch_bounds__(128) void qln_kernel(float* qraw,
                                                  const float* __restrict__ g,
                                                  const float* __restrict__ bb,
                                                  float* qsum) {
  const int wid  = blockIdx.x * 4 + (threadIdx.x >> 5);   // (b*T + t)*H + h
  const int lane = threadIdx.x & 31;
  const size_t base = (size_t)wid * 64;
  float v0 = qraw[base + lane];
  float v1 = qraw[base + lane + 32];
  float s  = v0 + v1;
  float sq = v0 * v0 + v1 * v1;
#pragma unroll
  for (int off = 16; off; off >>= 1) {
    s  += __shfl_xor(s,  off, 32);
    sq += __shfl_xor(sq, off, 32);
  }
  const float mean = s * (1.0f / 64.0f);
  const float var  = sq * (1.0f / 64.0f) - mean * mean;
  const float rstd = rsqrtf(var + LN_EPS_);
  const float y0 = (v0 - mean) * rstd * g[lane]      + bb[lane];
  const float y1 = (v1 - mean) * rstd * g[lane + 32] + bb[lane + 32];
  qraw[base + lane]      = y0;
  qraw[base + lane + 32] = y1;
  const int h = wid % H_;
  const int b = wid / (T_ * H_);
  atomicAdd(&qsum[((size_t)b * H_ + h) * 64 + lane],      y0);
  atomicAdd(&qsum[((size_t)b * H_ + h) * 64 + lane + 32], y1);
}

// ---- q finalize: subtract token mean, RoPE, write f16 (B,H,T,D) ------------
__global__ void qfin_kernel(const float* __restrict__ qln,
                            const float* __restrict__ qsum,
                            const float* __restrict__ cosp,
                            const float* __restrict__ sinp,
                            _Float16* __restrict__ q16) {
  const int idx = blockIdx.x * 256 + threadIdx.x;        // B*H*T*32 pairs
  if (idx >= B_ * H_ * T_ * 32) return;
  const int d2 = idx & 31;
  const int t  = (idx >> 5) % T_;
  const int h  = ((idx >> 5) / T_) % H_;
  const int b  = idx / (32 * T_ * H_);
  const int d  = 2 * d2;
  const size_t src = (((size_t)b * T_ + t) * H_ + h) * 64 + d;
  const size_t ms  = ((size_t)b * H_ + h) * 64 + d;
  const float invT = 1.0f / (float)T_;
  float x1 = qln[src]     - qsum[ms]     * invT;
  float x2 = qln[src + 1] - qsum[ms + 1] * invT;
  const float c0 = cosp[t * 64 + d],     s0 = sinp[t * 64 + d];
  const float c1 = cosp[t * 64 + d + 1], s1 = sinp[t * 64 + d + 1];
  const size_t dst = (((size_t)b * H_ + h) * T_ + t) * 64 + d;
  q16[dst]     = (_Float16)(x1 * c0 - x2 * s0);
  q16[dst + 1] = (_Float16)(x2 * c1 + x1 * s1);
}

// ---- kv prep: split kv_raw (B,T,KVH,2,D) -> RoPE(k), v as f16 (B,KVH,T,D) --
__global__ void kvprep_kernel(const float* __restrict__ kvraw,
                              const float* __restrict__ cosp,
                              const float* __restrict__ sinp,
                              _Float16* __restrict__ k16,
                              _Float16* __restrict__ v16) {
  const int idx = blockIdx.x * 256 + threadIdx.x;        // B*KVH*T*32 pairs
  if (idx >= B_ * KVH_ * T_ * 32) return;
  const int d2 = idx & 31;
  const int t  = (idx >> 5) % T_;
  const int g  = (idx >> 5) / T_;                        // b*KVH + kvh
  const int d  = 2 * d2;
  const size_t base =
      (((size_t)(g / KVH_) * T_ + t) * KVH_ + (g % KVH_)) * 128;
  const float k1 = kvraw[base + d],      k2 = kvraw[base + d + 1];
  const float w1 = kvraw[base + 64 + d], w2 = kvraw[base + 64 + d + 1];
  const float c0 = cosp[t * 64 + d],     s0 = sinp[t * 64 + d];
  const float c1 = cosp[t * 64 + d + 1], s1 = sinp[t * 64 + d + 1];
  const size_t dst = ((size_t)g * T_ + t) * 64 + d;
  k16[dst]     = (_Float16)(k1 * c0 - k2 * s0);
  k16[dst + 1] = (_Float16)(k2 * c1 + k1 * s1);
  v16[dst]     = (_Float16)w1;
  v16[dst + 1] = (_Float16)w2;
}

// ---- flash attention: 128 threads (4 waves), 64 queries/block --------------
// Wave w owns a 16-row query strip; S strip (16x64) in 4 f32 WMMA accums;
// online softmax stats in registers (16-lane shuffle reductions); P through a
// per-wave LDS strip. V staged via async-to-LDS; K staged with a VGPR
// transpose (ds_store_b64 of 4-row packs).
__global__ __launch_bounds__(128) void flash_attn_kernel(
    const _Float16* __restrict__ Q,   // (B,H,T,D) f16
    const _Float16* __restrict__ Kk,  // (B,KVH,T,D) f16
    const _Float16* __restrict__ Vv,  // (B,KVH,T,D) f16
    _Float16* __restrict__ O) {       // (B,T,H*D) f16
  __shared__ alignas(32) _Float16 Kt[64][64];        // K^T tile: [d][s]
  __shared__ alignas(32) _Float16 Vs[64][64];        // V tile:  [s][d]
  __shared__ alignas(32) _Float16 Ps[4][16][64];     // per-wave P strips

  const int tiles = T_ / 64;
  const int tile  = blockIdx.x % tiles;
  const int bh    = blockIdx.x / tiles;
  const int h     = bh % H_;
  const int b     = bh / H_;
  const int kvh   = (h >= 4) + (h >= 12);            // SPLITS (4,8,4)
  const int t0    = tile * 64;
  const int tid   = threadIdx.x;
  const int w     = tid >> 5;
  const int lane  = tid & 31;
  const int hi    = (lane >> 4) & 1;
  const int nl    = lane & 15;

  const size_t qoff = ((size_t)(b * H_ + h) * T_ + t0 + w * 16) * 64;
  v16h aq0 = wmma_frag_a(Q + qoff, 64);
  v16h aq1 = wmma_frag_a(Q + qoff + 32, 64);

  float mi[8], li[8];
  v8f oacc[4];
#pragma unroll
  for (int r = 0; r < 8; ++r) { mi[r] = -1e30f; li[r] = 0.0f; }
#pragma unroll
  for (int i = 0; i < 4; ++i)
#pragma unroll
    for (int j = 0; j < 8; ++j) oacc[i][j] = 0.0f;

  const size_t kvbase = (size_t)(b * KVH_ + kvh) * T_ * 64;

  for (int s0 = 0; s0 < T_; s0 += 64) {
    // --- V tile: async global->LDS copy (512 16B chunks / 128 threads) ---
#pragma unroll
    for (int i = 0; i < 4; ++i) {
      const int o   = tid + i * 128;
      const int row = o >> 3;
      const int d0  = (o & 7) * 8;
      async_copy16(&Vs[row][d0], Vv + kvbase + (size_t)(s0 + row) * 64 + d0);
    }
    // --- K tile: load 4 rows, transpose, ds_store_b64 packs ---
    {
      const int r4 = (tid >> 3) * 4;                 // 0..60
      const int d0 = (tid & 7) * 8;
      const _Float16* kp = Kk + kvbase + (size_t)(s0 + r4) * 64 + d0;
      h8 k0 = *(const h8*)(kp);
      h8 k1 = *(const h8*)(kp + 64);
      h8 k2 = *(const h8*)(kp + 128);
      h8 k3 = *(const h8*)(kp + 192);
#pragma unroll
      for (int j = 0; j < 8; ++j) {
        h4 t4 = {k0[j], k1[j], k2[j], k3[j]};
        *(h4*)&Kt[d0 + j][r4] = t4;
      }
    }
    // prefetch next tiles while this one is consumed
    if (s0 + 64 < T_) {
      __builtin_prefetch(Kk + kvbase + (size_t)(s0 + 64) * 64 + tid * 32, 0, 1);
      __builtin_prefetch(Vv + kvbase + (size_t)(s0 + 64) * 64 + tid * 32, 0, 1);
    }
    wait_async0();
    __syncthreads();

    // --- S strip = Q(16x64) * K^T(64x64) ---
    v8f sacc[4];
#pragma unroll
    for (int i = 0; i < 4; ++i)
#pragma unroll
      for (int j = 0; j < 8; ++j) sacc[i][j] = 0.0f;
#pragma unroll
    for (int kc = 0; kc < 64; kc += 32) {
      v16h vb[4];
#pragma unroll
      for (int nt = 0; nt < 4; ++nt)
        vb[nt] = wmma_frag_b(&Kt[0][0], 64, kc, nt * 16);
      const v16h aq = kc ? aq1 : aq0;
#pragma unroll
      for (int nt = 0; nt < 4; ++nt)
        sacc[nt] = wmma_mac(aq, vb[nt], sacc[nt]);
    }

    // --- online softmax per row ---
#pragma unroll
    for (int r = 0; r < 8; ++r) {
      float mloc = sacc[0][r] * SCALE_;
#pragma unroll
      for (int nt = 1; nt < 4; ++nt) mloc = fmaxf(mloc, sacc[nt][r] * SCALE_);
#pragma unroll
      for (int off = 8; off; off >>= 1) mloc = fmaxf(mloc, __shfl_xor(mloc, off, 32));
      const float mnew  = fmaxf(mi[r], mloc);
      const float alpha = __expf(mi[r] - mnew);
      float rs = 0.0f;
#pragma unroll
      for (int nt = 0; nt < 4; ++nt) {
        const float p = __expf(sacc[nt][r] * SCALE_ - mnew);
        Ps[w][r + hi * 8][nt * 16 + nl] = (_Float16)p;
        rs += p;
      }
#pragma unroll
      for (int off = 8; off; off >>= 1) rs += __shfl_xor(rs, off, 32);
      li[r] = li[r] * alpha + rs;
      mi[r] = mnew;
#pragma unroll
      for (int dt = 0; dt < 4; ++dt) oacc[dt][r] *= alpha;
    }

    // --- O strip += P(16x64) * V(64x64) ---
#pragma unroll
    for (int kc = 0; kc < 64; kc += 32) {
      v16h ap = wmma_frag_a(&Ps[w][0][kc], 64);
      v16h vb[4];
#pragma unroll
      for (int dt = 0; dt < 4; ++dt)
        vb[dt] = wmma_frag_b(&Vs[0][0], 64, kc, dt * 16);
#pragma unroll
      for (int dt = 0; dt < 4; ++dt)
        oacc[dt] = wmma_mac(ap, vb[dt], oacc[dt]);
    }
    __syncthreads();
  }

  // Epilogue: O /= l, f16 in (B,T,H*D) layout for the Wo GEMM.
#pragma unroll
  for (int r = 0; r < 8; ++r) {
    const float invl = 1.0f / li[r];
    const int trow = t0 + w * 16 + r + hi * 8;
#pragma unroll
    for (int dt = 0; dt < 4; ++dt) {
      const float ov = oacc[dt][r] * invl;
      O[((size_t)b * T_ + trow) * (H_ * D_) + h * 64 + dt * 16 + nl] = (_Float16)ov;
    }
  }
}

// ---------------------------------------------------------------------------
extern "C" void kernel_launch(void* const* d_in, const int* in_sizes, int n_in,
                              void* d_out, int out_size, void* d_ws, size_t ws_size,
                              hipStream_t stream) {
  (void)in_sizes; (void)n_in; (void)out_size; (void)ws_size;
  const float* x    = (const float*)d_in[0];
  const float* cosp = (const float*)d_in[1];
  const float* sinp = (const float*)d_in[2];
  const float* Wq   = (const float*)d_in[3];
  const float* Wkv  = (const float*)d_in[4];
  const float* Wo   = (const float*)d_in[5];
  const float* ln_g = (const float*)d_in[6];
  const float* ln_b = (const float*)d_in[7];
  float* out = (float*)d_out;

  char* ws = (char*)d_ws;
  size_t ofs = 0;
  auto alloc = [&](size_t bytes) -> void* {
    void* p = ws + ofs;
    ofs = (ofs + bytes + 255) & ~(size_t)255;
    return p;
  };
  const size_t NX  = (size_t)B_ * T_ * C_;            // 4M
  const size_t NKV = (size_t)B_ * T_ * KVH_ * 2 * D_; // 1.5M
  _Float16* x16   = (_Float16*)alloc(NX * 2);
  _Float16* Wq16  = (_Float16*)alloc((size_t)C_ * C_ * 2);
  _Float16* Wkv16 = (_Float16*)alloc((size_t)C_ * KVH_ * 2 * D_ * 2);
  _Float16* Wo16  = (_Float16*)alloc((size_t)C_ * C_ * 2);
  float*    qraw  = (float*)alloc(NX * 4);
  float*    kvraw = (float*)alloc(NKV * 4);
  float*    qsum  = (float*)alloc((size_t)B_ * H_ * D_ * 4);
  _Float16* q16   = (_Float16*)alloc(NX * 2);
  _Float16* k16   = (_Float16*)alloc((size_t)B_ * KVH_ * T_ * D_ * 2);
  _Float16* v16   = (_Float16*)alloc((size_t)B_ * KVH_ * T_ * D_ * 2);
  _Float16* att16 = (_Float16*)alloc(NX * 2);

  // 1. precision converts
  {
    int n = (int)NX;
    f32_to_f16_kernel<<<(n + 255) / 256, 256, 0, stream>>>(x, x16, n);
    n = C_ * C_;
    f32_to_f16_kernel<<<(n + 255) / 256, 256, 0, stream>>>(Wq, Wq16, n);
    n = C_ * KVH_ * 2 * D_;
    f32_to_f16_kernel<<<(n + 255) / 256, 256, 0, stream>>>(Wkv, Wkv16, n);
    n = C_ * C_;
    f32_to_f16_kernel<<<(n + 255) / 256, 256, 0, stream>>>(Wo, Wo16, n);
  }

  // 2. projection GEMMs (tile 128x64)
  gemm_f16_f32<<<dim3((B_ * T_) / 128, C_ / 64), 128, 0, stream>>>(
      x16, Wq16, qraw, B_ * T_, C_, C_);
  gemm_f16_f32<<<dim3((B_ * T_) / 128, (KVH_ * 2 * D_) / 64), 128, 0, stream>>>(
      x16, Wkv16, kvraw, B_ * T_, KVH_ * 2 * D_, C_);

  // 3. LayerNorm(q) + token sums
  (void)hipMemsetAsync(qsum, 0, (size_t)B_ * H_ * D_ * 4, stream);
  qln_kernel<<<(B_ * T_ * H_) / 4, 128, 0, stream>>>(qraw, ln_g, ln_b, qsum);

  // 4/5. q finalize + kv prep
  qfin_kernel<<<(B_ * H_ * T_ * 32) / 256, 256, 0, stream>>>(qraw, qsum, cosp, sinp, q16);
  kvprep_kernel<<<(B_ * KVH_ * T_ * 32 + 255) / 256, 256, 0, stream>>>(
      kvraw, cosp, sinp, k16, v16);

  // 6. flash attention
  flash_attn_kernel<<<B_ * H_ * (T_ / 64), 128, 0, stream>>>(q16, k16, v16, att16);

  // 7. output projection straight into d_out (f32)
  gemm_f16_f32<<<dim3((B_ * T_) / 128, C_ / 64), 128, 0, stream>>>(
      att16, Wo16, out, B_ * T_, C_, C_);
}
